// SpNCN_38474317037938
// MI455X (gfx1250) — matrix operand order repaired and checked by reading, not compiled, not run on hardware
//
#include <hip/hip_runtime.h>
#include <hip/hip_bf16.h>
#include <stdint.h>

// ---------------------------------------------------------------------------
// SpNCN on MI455X (gfx1250): 32 sequential steps, 2 fused kernels per step.
//  Kernel A: 4 independent GEMMs (bf16 WMMA, f32 acc) + fused LIF epilogues.
//  Kernel B: GEMM5 (s_h1 @ W0^T) + fused e_h0 epilogue.
// Block = 8 waves covering full M=256 for one 64-wide N group; the 64x128
// bf16 weight tile is staged in LDS via global_load_async_to_lds_b128 and
// reused by all 8 waves; each wave register-blocks 32x64 (8 WMMA accums).
// ---------------------------------------------------------------------------

#define B_   256
#define IN_  1024
#define H0_  2048
#define H1_  2048
#define LAB_ 10
#define LABP 16
#define WYPAD 64                   // W_y zero-padded to one 64-row N group
#define OUTW (IN_ + LAB_ + H0_)    // 3082

#define KCHUNK 128                 // K staged per LDS fill
#define LDSLD  136                 // 128 + 8 pad elements -> no bank conflicts

typedef __attribute__((ext_vector_type(16))) __bf16          v16bf;
typedef __attribute__((ext_vector_type(8)))  float           v8f;
typedef __attribute__((ext_vector_type(16))) unsigned short  v16u;
typedef int async_v4i __attribute__((vector_size(16)));      // builtin's pointee

#if __has_builtin(__builtin_amdgcn_global_load_async_to_lds_b128)
#define ASYNC_COPY 1
#else
#define ASYNC_COPY 0
#endif

__device__ __forceinline__ unsigned short f2bf(float f) {
  unsigned u = __float_as_uint(f);
  unsigned r = u + 0x7FFFu + ((u >> 16) & 1u);   // round-to-nearest-even
  return (unsigned short)(r >> 16);
}

// 16-byte global->LDS copy; async (ASYNCcnt, no VGPR round-trip) when available.
__device__ __forceinline__ void copy16(const unsigned short* g, unsigned short* l) {
#if ASYNC_COPY
  __builtin_amdgcn_global_load_async_to_lds_b128(
      (__attribute__((address_space(1))) async_v4i*)g,
      (__attribute__((address_space(3))) async_v4i*)l, 0, 0);
#else
  *(uint4*)l = *(const uint4*)g;
#endif
}

__device__ __forceinline__ void wait_copy() {
#if ASYNC_COPY
#if __has_builtin(__builtin_amdgcn_s_wait_asynccnt)
  __builtin_amdgcn_s_wait_asynccnt(0);
#else
  asm volatile("s_wait_asynccnt 0" ::: "memory");
#endif
#endif
}

// A fragment (16x32 MxK, 16-bit): lane l<16 -> row m0+l, K=k..k+15
//                                 lane l>=16 -> row m0+l-16, K=k+16..k+31
__device__ __forceinline__ v16bf load_a_frag(const unsigned short* base, int ld,
                                             int row0, int k, int lane) {
  const int r  = lane & 15;
  const int hi = lane >> 4;
  const unsigned short* p = base + (size_t)(row0 + r) * ld + (k + hi * 16);
  v16u f;
  *(uint4*)(&f)       = *(const uint4*)(p);
  *((uint4*)(&f) + 1) = *(const uint4*)(p + 8);
  return __builtin_bit_cast(v16bf, f);
}

// B fragment (32x16 KxN) from the LDS-staged 64xKCHUNK weight tile:
// lane l<16 -> W row n0loc+l, K=kk..kk+15; lane l>=16 -> same row, K=kk+16..kk+31.
__device__ __forceinline__ v16bf load_b_lds(const unsigned short* lb,
                                            int row0, int kk, int lane) {
  const int r  = lane & 15;
  const int hi = lane >> 4;
  const unsigned short* p = lb + (row0 + r) * LDSLD + kk + hi * 16;
  v16u f;
  *(uint4*)(&f)       = *(const uint4*)(p);
  *((uint4*)(&f) + 1) = *(const uint4*)(p + 8);
  return __builtin_bit_cast(v16bf, f);
}

__device__ __forceinline__ v8f wmma_bf16(v16bf a, v16bf b, v8f c) {
  return __builtin_amdgcn_wmma_f32_16x16x32_bf16(false, a, false, b,
                                                 (short)0, c, false, false);
}

// Block-cooperative GEMM: C[m0..m0+31][n0..n0+63] += A @ W^T for this wave,
// W tile staged through LDS by the whole block (256 threads).
__device__ __forceinline__ void block_gemm(const unsigned short* A, int ldA,
                                           const unsigned short* W, int ldW,
                                           unsigned short* lb,
                                           int m0, int n0, int K, int lane,
                                           v8f acc[2][4]) {
  for (int kc = 0; kc < K; kc += KCHUNK) {
    // ---- fill: 64 rows x 128 K bf16 = 16 KB, 4 x b128 per thread ----
    {
      const int t4 = (int)threadIdx.x * 4;
#pragma unroll
      for (int c = 0; c < 4; ++c) {
        const int ci  = t4 + c;          // 0..1023 chunks of 8 elements
        const int row = ci >> 4;         // 0..63
        const int ko  = (ci & 15) * 8;   // 0..120
        copy16(W + (size_t)(n0 + row) * ldW + kc + ko, lb + row * LDSLD + ko);
      }
    }
    wait_copy();
    __syncthreads();
    // ---- compute: 4 K-steps of 32, 8 WMMA each ----
#pragma unroll
    for (int kk = 0; kk < KCHUNK; kk += 32) {
      v16bf a0 = load_a_frag(A, ldA, m0,      kc + kk, lane);
      v16bf a1 = load_a_frag(A, ldA, m0 + 16, kc + kk, lane);
#pragma unroll
      for (int t = 0; t < 4; ++t) {
        v16bf b = load_b_lds(lb, t * 16, kk, lane);
        acc[0][t] = wmma_bf16(a0, b, acc[0][t]);
        acc[1][t] = wmma_bf16(a1, b, acc[1][t]);
      }
    }
    __syncthreads();
  }
}

// LIF: dt=1, tau_j=2, kappa=0.25, tau_m=20, gamma=R=1, thresh=0.4
__device__ __forceinline__ float lif_update(float I, float& v, float& j) {
  j = j + 0.5f * (I - 0.25f * j);
  v = v + 0.05f * (j - v);
  float s = (v > 0.4f) ? 1.0f : 0.0f;
  v = v * (1.0f - s);
  return s;
}

struct AParams {
  const unsigned short *sh0_r, *ein_r, *eh0_r;      // prev-step bf16 activations
  const unsigned short *Wx, *Wy, *Ex, *E0;          // bf16 weights (Wy padded)
  const float *x_data, *y_target, *eh0_f32;         // eh0_f32 = prev e_h0 (f32)
  float *inp_v, *inp_j, *lab_v, *lab_j;
  float *h0_v, *h0_j, *h0_x, *h1_v, *h1_j;
  unsigned short *sh0_w, *ein_w, *sh1_w;            // new bf16 activations
  float *out;
};

// block map: [0,16) inp | 16 lab | [17,49) h0 | [49,81) h1; wave w -> M rows 32w..
__global__ __launch_bounds__(256, 1) void spncn_step_a(AParams p) {
  const int bid  = blockIdx.x;
  const int lane = threadIdx.x & 31;
  const int hi   = lane >> 4;
  const int nlo  = lane & 15;
  const int m0   = (threadIdx.x >> 5) * 32;

  int layer, n0, K, ldA;
  const unsigned short *A, *W;
  if (bid < 16)      { layer = 0; n0 = bid * 64;        A = p.sh0_r; ldA = H0_; W = p.Wx; K = H0_; }
  else if (bid == 16){ layer = 1; n0 = 0;               A = p.sh0_r; ldA = H0_; W = p.Wy; K = H0_; }
  else if (bid < 49) { layer = 2; n0 = (bid - 17) * 64; A = p.ein_r; ldA = IN_; W = p.Ex; K = IN_; }
  else               { layer = 3; n0 = (bid - 49) * 64; A = p.eh0_r; ldA = H0_; W = p.E0; K = H0_; }

  __shared__ unsigned short lb[64 * LDSLD];

  v8f acc[2][4];
#pragma unroll
  for (int m = 0; m < 2; ++m)
#pragma unroll
    for (int t = 0; t < 4; ++t) acc[m][t] = v8f{0.f,0.f,0.f,0.f,0.f,0.f,0.f,0.f};

  block_gemm(A, ldA, W, K /*ldW == K for all weights*/, lb, m0, n0, K, lane, acc);

  if (layer == 0) {                      // inp: LIF + e_in = x_data - s
#pragma unroll
    for (int m = 0; m < 2; ++m)
#pragma unroll
      for (int t = 0; t < 4; ++t)
#pragma unroll
        for (int r = 0; r < 8; ++r) {
          const int row = m0 + m * 16 + r + hi * 8;
          const int col = n0 + 16 * t + nlo;
          const int idx = row * IN_ + col;
          float jj = p.inp_j[idx], vv = p.inp_v[idx];
          float s = lif_update(acc[m][t][r], vv, jj);
          p.inp_j[idx] = jj; p.inp_v[idx] = vv;
          float e = p.x_data[idx] - s;                 // alpha_gen = 1
          p.ein_w[idx] = f2bf(e);
          p.out[row * OUTW + col] = e;
        }
  } else if (layer == 1) {               // lab: only first 16-wide tile is real
#pragma unroll
    for (int m = 0; m < 2; ++m)
#pragma unroll
      for (int r = 0; r < 8; ++r) {
        const int row = m0 + m * 16 + r + hi * 8;
        const int col = nlo;
        const int idx = row * LABP + col;
        float jj = p.lab_j[idx], vv = p.lab_v[idx];
        float s = lif_update(acc[m][0][r], vv, jj);
        p.lab_j[idx] = jj; p.lab_v[idx] = vv;
        if (col < LAB_) {
          float e = p.y_target[row * LAB_ + col] - s;
          p.out[row * OUTW + IN_ + col] = e;
        }
      }
  } else if (layer == 2) {               // h0: I = e_in@E_x^T - e_h0_prev
#pragma unroll
    for (int m = 0; m < 2; ++m)
#pragma unroll
      for (int t = 0; t < 4; ++t)
#pragma unroll
        for (int r = 0; r < 8; ++r) {
          const int row = m0 + m * 16 + r + hi * 8;
          const int col = n0 + 16 * t + nlo;
          const int idx = row * H0_ + col;
          float I  = acc[m][t][r] - p.eh0_f32[idx];
          float jj = p.h0_j[idx], vv = p.h0_v[idx];
          float s = lif_update(I, vv, jj);
          p.h0_j[idx] = jj; p.h0_v[idx] = vv;
          p.h0_x[idx] = p.h0_x[idx] * 0.9f + s;       // trace, tau_tr = 10
          p.sh0_w[idx] = f2bf(s);
        }
  } else {                               // h1: I = e_h0 @ E0^T (e_h1 == 0)
#pragma unroll
    for (int m = 0; m < 2; ++m)
#pragma unroll
      for (int t = 0; t < 4; ++t)
#pragma unroll
        for (int r = 0; r < 8; ++r) {
          const int row = m0 + m * 16 + r + hi * 8;
          const int col = n0 + 16 * t + nlo;
          const int idx = row * H1_ + col;
          float jj = p.h1_j[idx], vv = p.h1_v[idx];
          float s = lif_update(acc[m][t][r], vv, jj);
          p.h1_j[idx] = jj; p.h1_v[idx] = vv;
          p.sh1_w[idx] = f2bf(s);
        }
  }
}

struct BParams {
  const unsigned short *sh1, *W0;
  const float *h0_x;
  float *eh0_f32;
  unsigned short *eh0_bf;
  float *out;
};

// e_h0 = h0_x - s_h1 @ W0^T   (256x2048x2048), 32 blocks x one 64-wide N group
__global__ __launch_bounds__(256, 1) void spncn_step_b(BParams p) {
  const int lane = threadIdx.x & 31;
  const int hi   = lane >> 4;
  const int nlo  = lane & 15;
  const int m0   = (threadIdx.x >> 5) * 32;
  const int n0   = blockIdx.x * 64;

  __shared__ unsigned short lb[64 * LDSLD];

  v8f acc[2][4];
#pragma unroll
  for (int m = 0; m < 2; ++m)
#pragma unroll
    for (int t = 0; t < 4; ++t) acc[m][t] = v8f{0.f,0.f,0.f,0.f,0.f,0.f,0.f,0.f};

  block_gemm(p.sh1, H1_, p.W0, H1_, lb, m0, n0, H1_, lane, acc);

#pragma unroll
  for (int m = 0; m < 2; ++m)
#pragma unroll
    for (int t = 0; t < 4; ++t)
#pragma unroll
      for (int r = 0; r < 8; ++r) {
        const int row = m0 + m * 16 + r + hi * 8;
        const int col = n0 + 16 * t + nlo;
        const int idx = row * H0_ + col;
        float e = p.h0_x[idx] - acc[m][t][r];
        p.eh0_f32[idx] = e;
        p.eh0_bf[idx]  = f2bf(e);
        p.out[row * OUTW + IN_ + LAB_ + col] = e;
      }
}

__global__ void zero_u4(uint4* p, long n) {
  long i = (long)blockIdx.x * blockDim.x + threadIdx.x;
  if (i < n) p[i] = make_uint4(0u, 0u, 0u, 0u);
}

__global__ void cvt_f32_bf16(const float* s, unsigned short* d, long n) {
  long i = (long)blockIdx.x * blockDim.x + threadIdx.x;
  if (i < n) d[i] = f2bf(s[i]);
}

extern "C" void kernel_launch(void* const* d_in, const int* in_sizes, int n_in,
                              void* d_out, int out_size, void* d_ws, size_t ws_size,
                              hipStream_t stream) {
  (void)in_sizes; (void)n_in; (void)out_size; (void)ws_size;
  const float* x_data   = (const float*)d_in[0];
  const float* y_target = (const float*)d_in[1];
  const float* W_x = (const float*)d_in[2];
  const float* E_x = (const float*)d_in[3];
  const float* W_y = (const float*)d_in[4];
  const float* W0  = (const float*)d_in[5];
  const float* E0  = (const float*)d_in[6];
  float* out = (float*)d_out;

  char* base = (char*)d_ws;
  size_t off = 0;
  auto alloc = [&](size_t bytes) -> char* {
    char* p = base + off;
    off = (off + bytes + 255) & ~(size_t)255;
    return p;
  };

  // ---- zeroed region (state + activations + padded W_y) ----
  float* inp_v = (float*)alloc((size_t)B_ * IN_ * 4);
  float* inp_j = (float*)alloc((size_t)B_ * IN_ * 4);
  float* lab_v = (float*)alloc((size_t)B_ * LABP * 4);
  float* lab_j = (float*)alloc((size_t)B_ * LABP * 4);
  float* h0_v  = (float*)alloc((size_t)B_ * H0_ * 4);
  float* h0_j  = (float*)alloc((size_t)B_ * H0_ * 4);
  float* h0_x  = (float*)alloc((size_t)B_ * H0_ * 4);
  float* h1_v  = (float*)alloc((size_t)B_ * H1_ * 4);
  float* h1_j  = (float*)alloc((size_t)B_ * H1_ * 4);
  float* eh0_f = (float*)alloc((size_t)B_ * H0_ * 4);
  unsigned short* sh0_a = (unsigned short*)alloc((size_t)B_ * H0_ * 2);
  unsigned short* sh0_b = (unsigned short*)alloc((size_t)B_ * H0_ * 2);
  unsigned short* ein_a = (unsigned short*)alloc((size_t)B_ * IN_ * 2);
  unsigned short* ein_b = (unsigned short*)alloc((size_t)B_ * IN_ * 2);
  unsigned short* eh0bf = (unsigned short*)alloc((size_t)B_ * H0_ * 2);
  unsigned short* sh1bf = (unsigned short*)alloc((size_t)B_ * H1_ * 2);
  unsigned short* Wy_bf = (unsigned short*)alloc((size_t)WYPAD * H0_ * 2);
  const size_t zero_bytes = off;
  // ---- bf16 weights ----
  unsigned short* Wx_bf = (unsigned short*)alloc((size_t)IN_ * H0_ * 2);
  unsigned short* Ex_bf = (unsigned short*)alloc((size_t)H0_ * IN_ * 2);
  unsigned short* E0_bf = (unsigned short*)alloc((size_t)H1_ * H0_ * 2);
  unsigned short* W0_bf = (unsigned short*)alloc((size_t)H0_ * H1_ * 2);

  const long zn = (long)(zero_bytes / 16);
  zero_u4<<<(unsigned)((zn + 255) / 256), 256, 0, stream>>>((uint4*)base, zn);

  auto cvt = [&](const float* s, unsigned short* d, long n) {
    cvt_f32_bf16<<<(unsigned)((n + 255) / 256), 256, 0, stream>>>(s, d, n);
  };
  cvt(W_x, Wx_bf, (long)IN_ * H0_);
  cvt(E_x, Ex_bf, (long)H0_ * IN_);
  cvt(E0,  E0_bf, (long)H1_ * H0_);
  cvt(W0,  W0_bf, (long)H0_ * H1_);
  cvt(W_y, Wy_bf, (long)LAB_ * H0_);   // rows 10..63 stay zero

  const int T = 32;  // reference setup: T = 32
  for (int t = 0; t < T; ++t) {
    const int pp = t & 1;
    AParams a;
    a.sh0_r = pp ? sh0_b : sh0_a;  a.sh0_w = pp ? sh0_a : sh0_b;
    a.ein_r = pp ? ein_b : ein_a;  a.ein_w = pp ? ein_a : ein_b;
    a.eh0_r = eh0bf;  a.eh0_f32 = eh0_f;
    a.Wx = Wx_bf; a.Wy = Wy_bf; a.Ex = Ex_bf; a.E0 = E0_bf;
    a.x_data = x_data; a.y_target = y_target;
    a.inp_v = inp_v; a.inp_j = inp_j; a.lab_v = lab_v; a.lab_j = lab_j;
    a.h0_v = h0_v; a.h0_j = h0_j; a.h0_x = h0_x;
    a.h1_v = h1_v; a.h1_j = h1_j;
    a.sh1_w = sh1bf; a.out = out;
    spncn_step_a<<<81, 256, 0, stream>>>(a);    // 16 inp + 1 lab + 32 h0 + 32 h1

    BParams b;
    b.sh1 = sh1bf; b.W0 = W0_bf; b.h0_x = h0_x;
    b.eh0_f32 = eh0_f; b.eh0_bf = eh0bf; b.out = out;
    spncn_step_b<<<32, 256, 0, stream>>>(b);
  }
}
